// Decoder_23467701305767
// MI455X (gfx1250) — matrix operand order repaired and checked by reading, not compiled, not run on hardware
//
#include <hip/hip_runtime.h>

typedef __attribute__((ext_vector_type(16))) __bf16 v16bf;
typedef __attribute__((ext_vector_type(8)))  __bf16 v8bf;
typedef __attribute__((ext_vector_type(8)))  float  v8f;

static constexpr int kT   = 500;
static constexpr int kMEL = 80;
static constexpr int kL   = 256;
static constexpr int kD   = 256;
static constexpr int TPB  = 512;            // 16 waves (wave32)
static constexpr size_t kMelOutElems = (size_t)128 * (kT * 2) * kMEL; // 10,240,000

struct DecParams {
  const float* dec_input;
  const float* enc;
  const float* pre_b1;  const float* pre_b2;
  const float* attn_bih; const float* attn_bhh;
  const float* proj1_b;
  const float* rnn1_bih; const float* rnn1_bhh;
  const float* rnn2_bih; const float* rnn2_bhh;
  const float* proj2_b;
  const __bf16 *Wpre1, *Wpre2, *Wattn_ih, *Wattn_hh, *Wproj1,
               *Wr1_ih, *Wr1_hh, *Wr2_ih, *Wr2_hh, *Wproj2;
  float* out;
};

__device__ __forceinline__ v8f vzero8() {
  v8f z;
#pragma unroll
  for (int i = 0; i < 8; ++i) z[i] = 0.0f;
  return z;
}

__device__ __forceinline__ v8f wmma_bf16(v16bf a, v16bf b, v8f c) {
  // D = A(16x32) * B(32x16) + C, fp32 accumulate
  return __builtin_amdgcn_wmma_f32_16x16x32_bf16(false, a, false, b, (short)0, c, false, false);
}

__device__ __forceinline__ v16bf pack16(v8bf lo, v8bf hi) {
  return __builtin_shufflevector(lo, hi, 0,1,2,3,4,5,6,7,8,9,10,11,12,13,14,15);
}

// A fragment (16x32 bf16) from row-major bf16 buffer (LDS), stride lda elems.
// ISA layout: lanes 0-15 hold K={0..7,16..23}, lanes 16-31 hold K={8..15,24..31}.
__device__ __forceinline__ v16bf load_a_bf16(const __bf16* src, int lda, int kbase, int lane) {
  const int row  = lane & 15;
  const int half = lane >> 4;
  const __bf16* p = src + row * lda + kbase + half * 8;
  v8bf g0 = *(const v8bf*)(p);
  v8bf g1 = *(const v8bf*)(p + 16);
  return pack16(g0, g1);
}

// A fragment from fp32 buffer with on-the-fly bf16 convert (hidden states live in fp32).
__device__ __forceinline__ v16bf load_a_f32(const float* src, int lda, int kbase, int lane) {
  const int row  = lane & 15;
  const int half = lane >> 4;
  const float* p = src + row * lda + kbase + half * 8;
  v16bf a;
#pragma unroll
  for (int i = 0; i < 8; ++i) a[i] = (__bf16)p[i];
#pragma unroll
  for (int i = 0; i < 8; ++i) a[8 + i] = (__bf16)p[16 + i];
  return a;
}

// B fragment (32x16): B[k][n] = W[n0+n][kbase+k], W row-major [N][ldk] bf16.
// ISA layout: lanes 0-15 (col n) hold K={0..7,16..23}, lanes 16-31 hold K={8..15,24..31}.
__device__ __forceinline__ v16bf load_b_w(const __bf16* W, int ldk, int n0, int kbase, int lane) {
  const int col  = lane & 15;
  const int half = lane >> 4;
  const __bf16* p = W + (size_t)(n0 + col) * ldk + kbase + half * 8;
  v8bf g0 = *(const v8bf*)(p);
  v8bf g1 = *(const v8bf*)(p + 16);
  return pack16(g0, g1);
}

__device__ __forceinline__ float sigm(float x) { return 1.0f / (1.0f + __expf(-x)); }

// Fused GRU N-tile: accumulate r/z/n gates over x-K and h-K, apply gates in registers.
// X: bf16 [16][256] LDS (first Kx cols valid); H: fp32 [16][256] LDS.
__device__ __forceinline__ v8f gru_tile(const __bf16* X, int Kx, const float* H,
                                        const __bf16* Wih, const __bf16* Whh,
                                        const float* bih, const float* bhh,
                                        int n0, int lane, int colc, int halfc) {
  v8f aR = vzero8(), aZ = vzero8(), aN = vzero8(), aHN = vzero8();
  for (int kb = 0; kb < Kx; kb += 32) {
    v16bf a = load_a_bf16(X, 256, kb, lane);
    aR = wmma_bf16(a, load_b_w(Wih, Kx,       n0, kb, lane), aR);
    aZ = wmma_bf16(a, load_b_w(Wih, Kx, 256 + n0, kb, lane), aZ);
    aN = wmma_bf16(a, load_b_w(Wih, Kx, 512 + n0, kb, lane), aN);
  }
  for (int kb = 0; kb < 256; kb += 32) {
    v16bf a = load_a_f32(H, 256, kb, lane);
    aR  = wmma_bf16(a, load_b_w(Whh, 256,       n0, kb, lane), aR);
    aZ  = wmma_bf16(a, load_b_w(Whh, 256, 256 + n0, kb, lane), aZ);
    aHN = wmma_bf16(a, load_b_w(Whh, 256, 512 + n0, kb, lane), aHN);
  }
  const int c = n0 + colc;
  const float br  = bih[c] + bhh[c];
  const float bz  = bih[256 + c] + bhh[256 + c];
  const float bin = bih[512 + c];
  const float bhn = bhh[512 + c];
  v8f hnew;
#pragma unroll
  for (int i = 0; i < 8; ++i) {
    float r = sigm(aR[i] + br);
    float z = sigm(aZ[i] + bz);
    float n = tanhf(aN[i] + bin + r * (aHN[i] + bhn));
    float hold = H[(halfc * 8 + i) * 256 + c];
    hnew[i] = (1.0f - z) * n + z * hold;
  }
  return hnew;
}

__global__ void cvt_pad_kernel(const float* __restrict__ src, __bf16* __restrict__ dst,
                               int N, int K, int Kp) {
  int i = blockIdx.x * 256 + threadIdx.x;
  if (i >= N * Kp) return;
  int n = i / Kp, k = i - n * Kp;
  dst[i] = (k < K) ? (__bf16)src[(size_t)n * K + k] : (__bf16)0.0f;
}

__global__ __launch_bounds__(TPB, 1) void decoder_kernel(DecParams P) {
  __shared__ float  sHa[16][256];   // attention GRU state (fp32)
  __shared__ float  sH1[16][256];   // rnn1 state
  __shared__ float  sH2[16][256];   // rnn2 state
  __shared__ __bf16 U[16][256];     // activation ping
  __shared__ __bf16 V[16][256];     // activation pong
  // total LDS = 3*16KB + 2*8KB = 64KB

  const int m0    = blockIdx.x * 16;     // batch tile base (8 blocks)
  const int tid   = threadIdx.x;
  const int lane  = tid & 31;
  const int w     = tid >> 5;            // wave id 0..15
  const int colc  = lane & 15;           // C-frag column
  const int halfc = lane >> 4;           // C-frag M-half

  for (int i = tid; i < 16 * 256; i += TPB) {
    (&sHa[0][0])[i] = 0.0f; (&sH1[0][0])[i] = 0.0f; (&sH2[0][0])[i] = 0.0f;
  }
  __syncthreads();

#pragma unroll 1
  for (int t = 0; t < kT; ++t) {
    // ---- frame -> U[:, 0:96) (zero-padded K for prenet1) ----
    for (int i = tid; i < 16 * 96; i += TPB) {
      int m = i / 96, k = i - m * 96;
      float v = (k < kMEL) ? P.dec_input[((size_t)(m0 + m) * kT + t) * kMEL + k] : 0.0f;
      U[m][k] = (__bf16)v;
    }
    __syncthreads();

    // ---- prenet1: x1 = relu(frame @ W1^T + b1) -> V   (N=256,K=96) ----
    {
      const int n0 = w * 16;
      v8f acc = vzero8();
      for (int kb = 0; kb < 96; kb += 32)
        acc = wmma_bf16(load_a_bf16(&U[0][0], 256, kb, lane),
                        load_b_w(P.Wpre1, 96, n0, kb, lane), acc);
      const float b = P.pre_b1[n0 + colc];
#pragma unroll
      for (int i = 0; i < 8; ++i)
        V[halfc * 8 + i][n0 + colc] = (__bf16)fmaxf(acc[i] + b, 0.0f);
    }
    __syncthreads();

    // ---- prenet2: x2 = relu(x1 @ W2^T + b2) -> U[:,0:128)  (N=128,K=256) ----
    {
      const int n0 = (w & 7) * 16;
      v8f acc = vzero8();
      if (w < 8) {
        for (int kb = 0; kb < 256; kb += 32)
          acc = wmma_bf16(load_a_bf16(&V[0][0], 256, kb, lane),
                          load_b_w(P.Wpre2, 256, n0, kb, lane), acc);
        const float b = P.pre_b2[n0 + colc];
#pragma unroll
        for (int i = 0; i < 8; ++i)
          U[halfc * 8 + i][n0 + colc] = (__bf16)fmaxf(acc[i] + b, 0.0f);
      }
    }
    __syncthreads();

    // ---- attention GRU: h_attn = GRU(x2, h_attn) -> sHa ----
    {
      const int n0 = w * 16;
      v8f hn = gru_tile(&U[0][0], 128, &sHa[0][0], P.Wattn_ih, P.Wattn_hh,
                        P.attn_bih, P.attn_bhh, n0, lane, colc, halfc);
      __syncthreads();              // all K-loops done before state overwrite
#pragma unroll
      for (int i = 0; i < 8; ++i)
        sHa[halfc * 8 + i][n0 + colc] = hn[i];
    }
    __syncthreads();

    // ---- dot attention + softmax (one wave per batch row) ----
    {
      const float* encb = P.enc + (size_t)(m0 + w) * kL * kD;
      const float4* h4 = (const float4*)(&sHa[w][0]);
      float s[8];
#pragma unroll
      for (int j = 0; j < 8; ++j) {
        const int l = j * 32 + lane;
        const float4* e4 = (const float4*)(encb + (size_t)l * kD);
        float acc = 0.0f;
        for (int q = 0; q < kD / 4; ++q) {
          float4 e = e4[q]; float4 h = h4[q];
          acc += e.x * h.x + e.y * h.y + e.z * h.z + e.w * h.w;
        }
        s[j] = acc;
      }
      float mx = s[0];
#pragma unroll
      for (int j = 1; j < 8; ++j) mx = fmaxf(mx, s[j]);
      for (int off = 16; off > 0; off >>= 1) mx = fmaxf(mx, __shfl_xor(mx, off, 32));
      float sum = 0.0f;
#pragma unroll
      for (int j = 0; j < 8; ++j) { s[j] = __expf(s[j] - mx); sum += s[j]; }
      for (int off = 16; off > 0; off >>= 1) sum += __shfl_xor(sum, off, 32);
      const float inv = 1.0f / sum;
      float* alignOut = P.out + kMelOutElems + (size_t)(m0 + w) * kL * kT + t;
#pragma unroll
      for (int j = 0; j < 8; ++j) {
        const int l = j * 32 + lane;
        const float pr = s[j] * inv;
        U[w][l] = (__bf16)pr;                 // stash p for ctx pass
        alignOut[(size_t)l * kT] = pr;        // alignment[b, l, t]
      }
      __syncthreads();
      // ctx[b, d] = sum_l p[l] * enc[b, l, d] ; lanes tile d
      const int d0 = lane * 8;
      float c[8] = {0,0,0,0,0,0,0,0};
      for (int l = 0; l < kL; ++l) {
        const float pr = (float)U[w][l];
        const float4* e4 = (const float4*)(encb + (size_t)l * kD + d0);
        float4 e0 = e4[0], e1 = e4[1];
        c[0] += pr * e0.x; c[1] += pr * e0.y; c[2] += pr * e0.z; c[3] += pr * e0.w;
        c[4] += pr * e1.x; c[5] += pr * e1.y; c[6] += pr * e1.z; c[7] += pr * e1.w;
      }
#pragma unroll
      for (int i = 0; i < 8; ++i) V[w][d0 + i] = (__bf16)c[i];
    }
    __syncthreads();

    // ---- proj1: dec_in = concat(ctx, h_attn) @ Wp1^T + b -> U  (N=256,K=512) ----
    {
      const int n0 = w * 16;
      v8f acc = vzero8();
      for (int kb = 0; kb < 512; kb += 32) {
        v16bf a = (kb < 256) ? load_a_bf16(&V[0][0], 256, kb, lane)
                             : load_a_f32(&sHa[0][0], 256, kb - 256, lane);
        acc = wmma_bf16(a, load_b_w(P.Wproj1, 512, n0, kb, lane), acc);
      }
      const float b = P.proj1_b[n0 + colc];
#pragma unroll
      for (int i = 0; i < 8; ++i)
        U[halfc * 8 + i][n0 + colc] = (__bf16)(acc[i] + b);
    }
    __syncthreads();

    // ---- rnn1: h1 = GRU(dec_in, h1); y1 = dec_in + h1 -> V ----
    {
      const int n0 = w * 16;
      v8f hn = gru_tile(&U[0][0], 256, &sH1[0][0], P.Wr1_ih, P.Wr1_hh,
                        P.rnn1_bih, P.rnn1_bhh, n0, lane, colc, halfc);
      __syncthreads();
#pragma unroll
      for (int i = 0; i < 8; ++i) {
        const int m = halfc * 8 + i;
        const float h = hn[i];
        sH1[m][n0 + colc] = h;
        V[m][n0 + colc] = (__bf16)((float)U[m][n0 + colc] + h);
      }
    }
    __syncthreads();

    // ---- rnn2: h2 = GRU(y1, h2); y2 = y1 + h2 -> U ----
    {
      const int n0 = w * 16;
      v8f hn = gru_tile(&V[0][0], 256, &sH2[0][0], P.Wr2_ih, P.Wr2_hh,
                        P.rnn2_bih, P.rnn2_bhh, n0, lane, colc, halfc);
      __syncthreads();
#pragma unroll
      for (int i = 0; i < 8; ++i) {
        const int m = halfc * 8 + i;
        const float h = hn[i];
        sH2[m][n0 + colc] = h;
        U[m][n0 + colc] = (__bf16)((float)V[m][n0 + colc] + h);
      }
    }
    __syncthreads();

    // ---- proj2: out = y2 @ Wp2^T + b -> d_out mel  (N=160,K=256) ----
    if (w < 10) {
      const int n0 = w * 16;
      v8f acc = vzero8();
      for (int kb = 0; kb < 256; kb += 32)
        acc = wmma_bf16(load_a_bf16(&U[0][0], 256, kb, lane),
                        load_b_w(P.Wproj2, 256, n0, kb, lane), acc);
      const float b = P.proj2_b[n0 + colc];
#pragma unroll
      for (int i = 0; i < 8; ++i) {
        const int m = halfc * 8 + i;
        P.out[(size_t)(m0 + m) * (kT * 2 * kMEL) + (size_t)t * (2 * kMEL) + n0 + colc] = acc[i] + b;
      }
    }
    __syncthreads();   // protect U before next step's frame store
  }
}

extern "C" void kernel_launch(void* const* d_in, const int* in_sizes, int n_in,
                              void* d_out, int out_size, void* d_ws, size_t ws_size,
                              hipStream_t stream) {
  (void)in_sizes; (void)n_in; (void)out_size; (void)ws_size;
  __bf16* wsb = (__bf16*)d_ws;
  size_t off = 0;
  auto alloc = [&](size_t n) { __bf16* p = wsb + off; off += n; return p; };

  __bf16* Wpre1    = alloc((size_t)256 * 96);
  __bf16* Wpre2    = alloc((size_t)128 * 256);
  __bf16* Wattn_ih = alloc((size_t)768 * 128);
  __bf16* Wattn_hh = alloc((size_t)768 * 256);
  __bf16* Wproj1   = alloc((size_t)256 * 512);
  __bf16* Wr1_ih   = alloc((size_t)768 * 256);
  __bf16* Wr1_hh   = alloc((size_t)768 * 256);
  __bf16* Wr2_ih   = alloc((size_t)768 * 256);
  __bf16* Wr2_hh   = alloc((size_t)768 * 256);
  __bf16* Wproj2   = alloc((size_t)160 * 256);

  struct Cvt { int idx; __bf16* dst; int N, K, Kp; } cv[10] = {
    {2,  Wpre1,    256, 80,  96},
    {4,  Wpre2,    128, 256, 256},
    {6,  Wattn_ih, 768, 128, 128},
    {7,  Wattn_hh, 768, 256, 256},
    {10, Wproj1,   256, 512, 512},
    {12, Wr1_ih,   768, 256, 256},
    {13, Wr1_hh,   768, 256, 256},
    {16, Wr2_ih,   768, 256, 256},
    {17, Wr2_hh,   768, 256, 256},
    {20, Wproj2,   160, 256, 256},
  };
  for (int i = 0; i < 10; ++i) {
    const int tot = cv[i].N * cv[i].Kp;
    cvt_pad_kernel<<<(tot + 255) / 256, 256, 0, stream>>>(
        (const float*)d_in[cv[i].idx], cv[i].dst, cv[i].N, cv[i].K, cv[i].Kp);
  }

  DecParams P;
  P.dec_input = (const float*)d_in[0];
  P.enc       = (const float*)d_in[1];
  P.pre_b1    = (const float*)d_in[3];
  P.pre_b2    = (const float*)d_in[5];
  P.attn_bih  = (const float*)d_in[8];
  P.attn_bhh  = (const float*)d_in[9];
  P.proj1_b   = (const float*)d_in[11];
  P.rnn1_bih  = (const float*)d_in[14];
  P.rnn1_bhh  = (const float*)d_in[15];
  P.rnn2_bih  = (const float*)d_in[18];
  P.rnn2_bhh  = (const float*)d_in[19];
  P.proj2_b   = (const float*)d_in[21];
  P.Wpre1 = Wpre1; P.Wpre2 = Wpre2;
  P.Wattn_ih = Wattn_ih; P.Wattn_hh = Wattn_hh;
  P.Wproj1 = Wproj1;
  P.Wr1_ih = Wr1_ih; P.Wr1_hh = Wr1_hh;
  P.Wr2_ih = Wr2_ih; P.Wr2_hh = Wr2_hh;
  P.Wproj2 = Wproj2;
  P.out = (float*)d_out;

  decoder_kernel<<<dim3(8), dim3(TPB), 0, stream>>>(P);
}
